// twolayerGCNc_44521630990796
// MI455X (gfx1250) — matrix operand order repaired and checked by reading, not compiled, not run on hardware
//
#include <hip/hip_runtime.h>

typedef __attribute__((ext_vector_type(16))) __bf16 v16bf;
typedef __attribute__((ext_vector_type(8)))  float  v8f;

#define C_SELF 1.0f

// ---------------------------------------------------------------- utilities
__global__ void zero_f32_kernel(float* __restrict__ p, long long n) {
  long long i = (long long)blockIdx.x * blockDim.x + threadIdx.x;
  long long s = (long long)gridDim.x * blockDim.x;
  for (; i < n; i += s) p[i] = 0.0f;
}

__global__ void deg_kernel(const int* __restrict__ row, float* __restrict__ deg, int E) {
  int i = blockIdx.x * blockDim.x + threadIdx.x;
  int s = gridDim.x * blockDim.x;
  for (; i < E; i += s) atomicAdd(&deg[row[i]], 1.0f);
}

__global__ void dis_kernel(const float* __restrict__ deg, float* __restrict__ dis, int Nn) {
  int i = blockIdx.x * blockDim.x + threadIdx.x;
  int s = gridDim.x * blockDim.x;
  for (; i < Nn; i += s) {
    float d = deg[i] + C_SELF;               // deg includes self-loop weight c
    dis[i] = (d > 0.0f) ? rsqrtf(d) : 0.0f;  // deg^{-1/2}
  }
}

__global__ void norm_kernel(const int* __restrict__ row, const int* __restrict__ col,
                            const float* __restrict__ dis, float* __restrict__ nrm, int E) {
  int i = blockIdx.x * blockDim.x + threadIdx.x;
  int s = gridDim.x * blockDim.x;
  for (; i < E; i += s) nrm[i] = dis[row[i]] * dis[col[i]];
}

// Repack W[K,N] (f32, row-major) into WMMA B-fragment order (bf16):
// frag index = ((kt*NT + nt)*32 + lane)*16 + j ; k = kt*32 + (lane>>4)*16 + j ;
// n = nt*16 + (lane&15).  Zero-pad n >= N (for N=40 -> Npad=48).
__global__ void pack_b_kernel(const float* __restrict__ W, __bf16* __restrict__ out,
                              int K, int N, int Npad) {
  const int ntiles = Npad / 16;
  const long long total = (long long)(K / 32) * ntiles * 512;
  long long idx = (long long)blockIdx.x * blockDim.x + threadIdx.x;
  long long s = (long long)gridDim.x * blockDim.x;
  for (; idx < total; idx += s) {
    int perKt = ntiles * 512;
    int t  = (int)(idx / perKt);
    int r  = (int)(idx % perKt);
    int u  = r / 512;
    int r2 = r % 512;
    int lane = r2 >> 4;
    int j    = r2 & 15;
    int k = t * 32 + ((lane >> 4) << 4) + j;
    int n = u * 16 + (lane & 15);
    float v = (k < K && n < N) ? W[(size_t)k * N + n] : 0.0f;
    out[idx] = (__bf16)v;
  }
}

// ---------------------------------------------------------------- WMMA GEMM
// C[M, Nvalid] = A[M, KT*32] @ Bpacked ;  f32 in, bf16 WMMA, f32 accumulate.
// Block = 256 threads = 8 waves; each wave computes a 16-row strip x NT tiles.
// The WHOLE packed B (KT*NT*1024 bytes) is staged to LDS once at kernel start
// via global_load_async_to_lds_b128 (ASYNCcnt path) -> no per-k barriers.
template <int KT, int NT>
__global__ void __launch_bounds__(256)
gemm_bf16_wmma_kernel(const float* __restrict__ A, const __bf16* __restrict__ Bp,
                      float* __restrict__ C, int M, int Nvalid) {
  constexpr int K = KT * 32;
  __shared__ v16bf sB[KT * NT * 32];        // full packed B: 64 KB (L1), 12 KB (L2)

  const int tid  = threadIdx.x;
  const int lane = tid & 31;
  const int wave = tid >> 5;

  // ---- async stage of all of B into LDS (b128 per op, exact multiple of 256)
  {
    constexpr int NV = KT * NT * 64;        // number of 16-byte vectors
    constexpr int PER = NV / 256;           // per-thread (16 for L1, 3 for L2)
    static_assert(NV % 256 == 0, "B size must tile the block");
    unsigned sbase = (unsigned)(uintptr_t)(&sB[0]);
    unsigned long long gbase = (unsigned long long)(uintptr_t)Bp;
#pragma unroll
    for (int p = 0; p < PER; ++p) {
      int i = tid + p * 256;
      unsigned      l = sbase + (unsigned)i * 16u;
      unsigned long long g = gbase + (unsigned long long)i * 16ull;
      asm volatile("global_load_async_to_lds_b128 %0, %1, off"
                   :: "v"(l), "v"(g) : "memory");
    }
    asm volatile("s_wait_asynccnt 0x0" ::: "memory");
  }
  __syncthreads();                          // B visible to all waves

  const int rowTiles = (M + 15) >> 4;
  const int rowTile  = blockIdx.x * (blockDim.x >> 5) + wave;
  const bool valid = rowTile < rowTiles;

  int row = rowTile * 16 + (lane & 15);
  if (row > M - 1) row = M - 1;              // clamp so tail waves load in-bounds
  const float* arow = A + (size_t)row * K;
  const int kbL = (lane >> 4) << 4;          // 0 or 16 (A fragment K base in tile)

  v8f acc[NT];
#pragma unroll
  for (int u = 0; u < NT; ++u)
#pragma unroll
    for (int i = 0; i < 8; ++i) acc[u][i] = 0.0f;

#pragma unroll
  for (int kt = 0; kt < KT; ++kt) {
    // A fragment: one contiguous 64 B f32 chunk per lane, cvt -> bf16
    const int k0 = kt * 32 + kbL;
    float4 f0 = *(const float4*)(arow + k0);
    float4 f1 = *(const float4*)(arow + k0 + 4);
    float4 f2 = *(const float4*)(arow + k0 + 8);
    float4 f3 = *(const float4*)(arow + k0 + 12);
    if (kt + 1 < KT) {                        // global_prefetch_b8 of next tile
      __builtin_prefetch(arow + k0 + 32, 0, 1);
      __builtin_prefetch(arow + k0 + 48, 0, 1);
    }
    v16bf a;
    a[0]=(__bf16)f0.x;  a[1]=(__bf16)f0.y;  a[2]=(__bf16)f0.z;  a[3]=(__bf16)f0.w;
    a[4]=(__bf16)f1.x;  a[5]=(__bf16)f1.y;  a[6]=(__bf16)f1.z;  a[7]=(__bf16)f1.w;
    a[8]=(__bf16)f2.x;  a[9]=(__bf16)f2.y;  a[10]=(__bf16)f2.z; a[11]=(__bf16)f2.w;
    a[12]=(__bf16)f3.x; a[13]=(__bf16)f3.y; a[14]=(__bf16)f3.z; a[15]=(__bf16)f3.w;

#pragma unroll
    for (int u = 0; u < NT; ++u) {
      v16bf b = sB[(kt * NT + u) * 32 + lane];
      acc[u] = __builtin_amdgcn_wmma_f32_16x16x32_bf16(
          /*neg_a=*/false, a, /*neg_b=*/false, b,
          /*c_mod=*/(short)0, acc[u], /*reuse_a=*/false, /*reuse_b=*/false);
    }
  }

  // Store: VGPR v -> M = v + 8*(lane>=16); N = lane&15 within tile
  const int col0  = lane & 15;
  const int mBase = rowTile * 16 + ((lane >> 4) << 3);
#pragma unroll
  for (int u = 0; u < NT; ++u) {
    int c = u * 16 + col0;
    if (valid && c < Nvalid) {
#pragma unroll
      for (int v = 0; v < 8; ++v)
        C[(size_t)(mBase + v) * Nvalid + c] = acc[u][v];
    }
  }
}

// ------------------------------------------------------- edge scatter-add
// out[row[e], :] += norm[e] * H[col[e], :]  (one wave per edge, lanes stripe F)
__global__ void agg_kernel(const float* __restrict__ H, const float* __restrict__ nrm,
                           const int* __restrict__ row, const int* __restrict__ col,
                           float* __restrict__ out, int E, int F) {
  int gid  = blockIdx.x * blockDim.x + threadIdx.x;
  int wav  = gid >> 5;
  int lane = gid & 31;
  int nw   = (gridDim.x * blockDim.x) >> 5;
  if (F == 128) {                           // fast path: float4 gather per lane
    for (int e = wav; e < E; e += nw) {
      int r = row[e], c = col[e];
      float w = nrm[e];
      const float4 v = *(const float4*)(H + (size_t)c * 128 + lane * 4);
      float* o = out + (size_t)r * 128 + lane * 4;
      atomicAdd(o + 0, w * v.x);
      atomicAdd(o + 1, w * v.y);
      atomicAdd(o + 2, w * v.z);
      atomicAdd(o + 3, w * v.w);
    }
  } else {
    for (int e = wav; e < E; e += nw) {
      int r = row[e], c = col[e];
      float w = nrm[e];
      const float* hc = H + (size_t)c * F;
      float* orow = out + (size_t)r * F;
      for (int f = lane; f < F; f += 32) atomicAdd(&orow[f], w * hc[f]);
    }
  }
}

// out = [relu]( out + c*dis_i^2 * H + b_f )
__global__ void finalize_kernel(float* __restrict__ out, const float* __restrict__ H,
                                const float* __restrict__ dis, const float* __restrict__ b,
                                int Nn, int F, int doRelu) {
  long long total = (long long)Nn * F;
  long long i = (long long)blockIdx.x * blockDim.x + threadIdx.x;
  long long s = (long long)gridDim.x * blockDim.x;
  for (; i < total; i += s) {
    int node = (int)(i / F);
    int f    = (int)(i % F);
    float d = dis[node];
    float v = out[i] + C_SELF * d * d * H[i] + b[f];
    out[i] = doRelu ? fmaxf(v, 0.0f) : v;
  }
}

// ---------------------------------------------------------------- launcher
extern "C" void kernel_launch(void* const* d_in, const int* in_sizes, int n_in,
                              void* d_out, int out_size, void* d_ws, size_t ws_size,
                              hipStream_t stream) {
  const float* x  = (const float*)d_in[0];
  const int*   ei = (const int*)d_in[1];
  const float* W1 = (const float*)d_in[2];
  const float* b1 = (const float*)d_in[3];
  const float* W2 = (const float*)d_in[4];
  const float* b2 = (const float*)d_in[5];

  const int F0 = 256, FH = 128, FL = 40, FLpad = 48;
  const int Nn = in_sizes[0] / F0;
  const int E  = in_sizes[1] / 2;
  const int* row = ei;
  const int* col = ei + E;

  char* base = (char*)d_ws;
  size_t off = 0;
  auto carve = [&](size_t bytes) -> void* {
    void* p = base + off;
    off = (off + bytes + 255) & ~(size_t)255;
    return p;
  };
  float*  deg = (float*)carve((size_t)Nn * 4);
  float*  dis = (float*)carve((size_t)Nn * 4);
  float*  nrm = (float*)carve((size_t)E * 4);
  float*  H1  = (float*)carve((size_t)Nn * FH * 4);
  float*  A1  = (float*)carve((size_t)Nn * FH * 4);
  float*  H2  = (float*)carve((size_t)Nn * FL * 4);
  __bf16* W1p = (__bf16*)carve((size_t)(F0 / 32) * (FH / 16) * 512 * 2);
  __bf16* W2p = (__bf16*)carve((size_t)(FH / 32) * (FLpad / 16) * 512 * 2);
  float*  out = (float*)d_out;

  auto gsz = [](long long n) { long long b = (n + 255) / 256; return (int)(b < 4096 ? b : 4096); };

  // --- degree / normalization prep (shared by both layers)
  zero_f32_kernel<<<gsz(Nn), 256, 0, stream>>>(deg, Nn);
  deg_kernel<<<gsz(E), 256, 0, stream>>>(row, deg, E);
  dis_kernel<<<gsz(Nn), 256, 0, stream>>>(deg, dis, Nn);
  norm_kernel<<<gsz(E), 256, 0, stream>>>(row, col, dis, nrm, E);

  // --- weight repack to WMMA B-fragment bf16 layout
  pack_b_kernel<<<gsz((long long)(F0/32)*(FH/16)*512), 256, 0, stream>>>(W1, W1p, F0, FH, FH);
  pack_b_kernel<<<gsz((long long)(FH/32)*(FLpad/16)*512), 256, 0, stream>>>(W2, W2p, FH, FL, FLpad);

  // --- layer 1: H1 = x @ W1 ; aggregate ; self-loop + bias + relu
  {
    int rowTiles = (Nn + 15) / 16;
    int blocks = (rowTiles + 7) / 8;
    gemm_bf16_wmma_kernel<8, 8><<<blocks, 256, 0, stream>>>(x, W1p, H1, Nn, FH);
  }
  zero_f32_kernel<<<gsz((long long)Nn * FH), 256, 0, stream>>>(A1, (long long)Nn * FH);
  agg_kernel<<<2048, 256, 0, stream>>>(H1, nrm, row, col, A1, E, FH);
  finalize_kernel<<<gsz((long long)Nn * FH), 256, 0, stream>>>(A1, H1, dis, b1, Nn, FH, 1);

  // --- layer 2: H2 = A1 @ W2 ; aggregate into d_out ; self-loop + bias
  {
    int rowTiles = (Nn + 15) / 16;
    int blocks = (rowTiles + 7) / 8;
    gemm_bf16_wmma_kernel<4, 3><<<blocks, 256, 0, stream>>>(A1, W2p, H2, Nn, FL);
  }
  zero_f32_kernel<<<gsz((long long)Nn * FL), 256, 0, stream>>>(out, (long long)Nn * FL);
  agg_kernel<<<2048, 256, 0, stream>>>(H2, nrm, row, col, out, E, FL);
  finalize_kernel<<<gsz((long long)Nn * FL), 256, 0, stream>>>(out, H2, dis, b2, Nn, FL, 0);

  (void)n_in; (void)out_size; (void)ws_size;
}
